// ADIAModel_56667798503527
// MI455X (gfx1250) — matrix operand order repaired and checked by reading, not compiled, not run on hardware
//
#include <hip/hip_runtime.h>
#include <math.h>

typedef _Float16 v8h  __attribute__((ext_vector_type(8)));
typedef _Float16 v16h __attribute__((ext_vector_type(16)));
typedef float    v8f  __attribute__((ext_vector_type(8)));

#define B_   8
#define E_   240
#define M_   (B_*E_)
#define C_   5
#define N_   256
#define D_   64
#define H_   4
#define HD_  16
#define NB_  5
#define NL_  2

// ---- LDS layout for the conv-stack kernel (dynamic LDS; 320KB/WGP on CDNA5) ----
#define HB_STRIDE   257          // f32 activation row stride (odd -> conflict free)
#define H16_STRIDE  72           // halves per padded column (=36 dwords, 16B aligned)
#define HB_FLOATS   (64*HB_STRIDE)
#define CB_FLOATS   (64*HB_STRIDE)
#define H16_HALVES  (258*H16_STRIDE)   // cols -1..256 (zero halo)
#define W16_HALVES  (192*H16_STRIDE)   // [tap*64+dout][din]
#define CONV_LDS_BYTES (HB_FLOATS*4 + CB_FLOATS*4 + H16_HALVES*2 + W16_HALVES*2 + 512*4 + 16*4)

__device__ __forceinline__ float gelu_exact(float x) {
    return 0.5f * x * (1.0f + erff(x * 0.70710678118654752440f));
}

// power-of-two blockDim tree reduction; returns sum to all threads
__device__ __forceinline__ float block_reduce_sum(float v, float* buf) {
    const int t = threadIdx.x;
    buf[t] = v;
    __syncthreads();
    for (int off = blockDim.x >> 1; off > 0; off >>= 1) {
        if (t < off) buf[t] += buf[t + off];
        __syncthreads();
    }
    const float s = buf[0];
    __syncthreads();
    return s;
}

// ============================================================================
// Kernel 1: stem linear + 5x (conv1d k=3 -> GroupNorm(8) -> GELU -> residual)
//           + avg-pool.  One workgroup per sample m (B*E=1920), 256 thr = 8 waves.
//           Conv GEMM runs on v_wmma_f32_16x16x32_f16 (f32 accum).
// ============================================================================
__global__ __launch_bounds__(256) void conv_stack_kernel(
    const float* __restrict__ edge_data, const float* __restrict__ stem_w,
    const float* __restrict__ stem_b,    const float* __restrict__ conv_w,
    const float* __restrict__ gn_g,      const float* __restrict__ gn_b,
    float* __restrict__ emb)
{
    extern __shared__ char smem[];
    float*    hbuf = (float*)smem;                    // f32 h [64][257]
    float*    cbuf = hbuf + HB_FLOATS;                // f32 conv out [64][257]
    _Float16* h16  = (_Float16*)(cbuf + CB_FLOATS);   // f16 h^T [258 cols][72]
    _Float16* w16  = h16 + H16_HALVES;                // f16 weights [192][72]
    float*    red  = (float*)(w16 + W16_HALVES);      // 512 f32 reduce scratch
    float*    mv   = red + 512;                       // 8 x (mean, rstd)

    const int m    = blockIdx.x;
    const int t    = threadIdx.x;
    const int lane = t & 31;
    const int wv   = t >> 5;
    const int lm   = lane & 15;        // row/col within tile
    const int hi16 = lane >> 4;        // upper half-wave?
    const int r    = wv & 3;           // output-row tile block (0..3)
    const int cs   = wv >> 2;          // column half (0..1)

    // ---- stem: h[d][n] = sum_c x[c][n]*stem_w[c][d] + stem_b[d] ----
    {
        float* swf = (float*)w16;                     // reuse as f32 scratch
        if (t < C_*D_) swf[t] = stem_w[t];
        if (t < D_)    swf[C_*D_ + t] = stem_b[t];
        __syncthreads();
        const float* xg = edge_data + (size_t)m * (C_*N_);
        float xc[C_];
        #pragma unroll
        for (int c = 0; c < C_; ++c) xc[c] = xg[c*N_ + t];   // t = column n
        for (int d = 0; d < D_; ++d) {
            float acc = swf[C_*D_ + d];
            #pragma unroll
            for (int c = 0; c < C_; ++c) acc += xc[c] * swf[c*D_ + d];
            hbuf[d*HB_STRIDE + t]     = acc;
            h16[(t+1)*H16_STRIDE + d] = (_Float16)acc;       // padded col = n+1
        }
        if (t < D_) {                                        // zero halo columns
            h16[0*H16_STRIDE + t]   = (_Float16)0.0f;
            h16[257*H16_STRIDE + t] = (_Float16)0.0f;
        }
    }
    __syncthreads();

    for (int blk = 0; blk < NB_; ++blk) {
        // stage conv weights: w16[(k*64+dout)*72 + din] = conv_w[blk][dout][din][k]
        const float* wg = conv_w + (size_t)blk * (D_*D_*3);
        for (int idx = t; idx < D_*D_*3; idx += 256) {
            const int dout = idx / (D_*3);
            const int rem  = idx - dout*(D_*3);
            const int din  = rem / 3;
            const int k    = rem - din*3;
            w16[(k*D_ + dout)*H16_STRIDE + din] = (_Float16)wg[idx];
        }
        __syncthreads();

        // ---- conv as 3 shifted GEMMs on WMMA; each wave: 8 tiles x 6 K-steps ----
        v8f acc[8] = {};
        const int abase = hi16 ? 8 : 0;     // ISA A-layout K base for upper lanes
        const int koff  = hi16 ? 8 : 0;     // ISA B-layout K base for upper lanes
        #pragma unroll
        for (int k = 0; k < 3; ++k) {
            #pragma unroll
            for (int e2 = 0; e2 < 2; ++e2) {
                const int e0 = e2 * 32;
                // A fragment: weights row dout = r*16+lm, K pairs at base and base+16
                const _Float16* ap = &w16[(k*D_ + r*16 + lm)*H16_STRIDE + e0 + abase];
                const v8h alo = *(const v8h*)(ap);
                const v8h ahi = *(const v8h*)(ap + 16);
                const v16h a = __builtin_shufflevector(alo, ahi,
                    0,1,2,3,4,5,6,7,8,9,10,11,12,13,14,15);
                #pragma unroll
                for (int cb = 0; cb < 8; ++cb) {
                    // B fragment: column n0+lm, K=din pairs at base and base+16
                    const int col = cs*128 + cb*16 + lm + k;   // = n + (k-1) + 1 pad
                    const _Float16* bp = &h16[col*H16_STRIDE + e0 + koff];
                    const v8h blo = *(const v8h*)(bp);
                    const v8h bhi = *(const v8h*)(bp + 16);
                    const v16h bf = __builtin_shufflevector(blo, bhi,
                        0,1,2,3,4,5,6,7,8,9,10,11,12,13,14,15);
                    acc[cb] = __builtin_amdgcn_wmma_f32_16x16x32_f16(
                        false, a, false, bf, (short)0, acc[cb], false, false);
                }
            }
        }
        // store C tiles (VGPR j -> row j / j+8 per ISA C layout)
        #pragma unroll
        for (int cb = 0; cb < 8; ++cb) {
            const int n0 = cs*128 + cb*16;
            #pragma unroll
            for (int j = 0; j < 8; ++j) {
                const int row = r*16 + j + (hi16 ? 8 : 0);
                cbuf[row*HB_STRIDE + n0 + lm] = acc[cb][j];
            }
        }
        __syncthreads();

        // ---- GroupNorm stats: 8 groups x (8ch x 256) ----
        {
            const int g  = t >> 5;
            const int sl = t & 31;
            float s = 0.f, s2 = 0.f;
            const int base = sl * 64;
            for (int i = 0; i < 64; ++i) {
                const int flat = base + i;
                const int ch   = g*8 + (flat >> 8);
                const int col  = flat & 255;
                const float v  = cbuf[ch*HB_STRIDE + col];
                s += v; s2 += v*v;
            }
            red[t] = s; red[256 + t] = s2;
            __syncthreads();
            if (sl == 0) {
                float ts = 0.f, ts2 = 0.f;
                for (int i = 0; i < 32; ++i) { ts += red[g*32+i]; ts2 += red[256+g*32+i]; }
                const float mean = ts * (1.0f/2048.0f);
                const float var  = ts2 * (1.0f/2048.0f) - mean*mean;
                mv[g*2]   = mean;
                mv[g*2+1] = rsqrtf(var + 1e-5f);
            }
            __syncthreads();
        }
        // ---- apply: h += gelu(gn(c)); refresh f16 copy ----
        {
            const int d0  = t >> 2;
            const int c0  = (t & 3) * 64;
            const int grp = d0 >> 3;
            const float mean = mv[grp*2];
            const float rstd = mv[grp*2+1];
            const float gain = gn_g[blk*D_ + d0];
            const float bias = gn_b[blk*D_ + d0];
            for (int i = 0; i < 64; ++i) {
                const int col = c0 + i;
                const float c = cbuf[d0*HB_STRIDE + col];
                const float y = (c - mean) * rstd * gain + bias;
                const float nh = hbuf[d0*HB_STRIDE + col] + gelu_exact(y);
                hbuf[d0*HB_STRIDE + col]     = nh;
                h16[(col+1)*H16_STRIDE + d0] = (_Float16)nh;
            }
        }
        __syncthreads();
    }

    // ---- average pool over N ----
    {
        const int d0 = t >> 2;
        const int c0 = (t & 3) * 64;
        float s = 0.f;
        for (int i = 0; i < 64; ++i) s += hbuf[d0*HB_STRIDE + c0 + i];
        red[t] = s;
        __syncthreads();
        if (t < D_) {
            const float tot = red[t*4] + red[t*4+1] + red[t*4+2] + red[t*4+3];
            emb[(size_t)m*D_ + t] = tot * (1.0f/256.0f);
        }
    }
}

// ============================================================================
// Kernel 2: type-embedding merge: gelu(LN(concat(emb,temb) @ em_w + em_b))
// ============================================================================
__global__ __launch_bounds__(64) void merge_kernel(
    const int* __restrict__ edge_types, const float* __restrict__ type_emb,
    const float* __restrict__ em_w, const float* __restrict__ em_b,
    const float* __restrict__ em_g, const float* __restrict__ em_bt,
    float* __restrict__ emb)
{
    __shared__ float cat[128];
    __shared__ float buf[64];
    const int m = blockIdx.x, t = threadIdx.x;
    cat[t]      = emb[(size_t)m*64 + t];
    cat[64 + t] = type_emb[edge_types[m]*64 + t];
    __syncthreads();
    float a = em_b[t];
    for (int j = 0; j < 128; ++j) a += cat[j] * em_w[j*64 + t];
    const float mean = block_reduce_sum(a, buf) * (1.0f/64.0f);
    const float dv   = a - mean;
    const float var  = block_reduce_sum(dv*dv, buf) * (1.0f/64.0f);
    const float y    = dv * rsqrtf(var + 1e-5f) * em_g[t] + em_bt[t];
    emb[(size_t)m*64 + t] = gelu_exact(y);
}

// ============================================================================
// Transformer layer pieces
// ============================================================================
__global__ __launch_bounds__(192) void qkv_kernel(
    const float* __restrict__ emb, const float* __restrict__ wqkv,
    const float* __restrict__ bqkv, float* __restrict__ qkv)
{
    __shared__ float row[64];
    const int m = blockIdx.x, t = threadIdx.x;
    if (t < 64) row[t] = emb[(size_t)m*64 + t];
    __syncthreads();
    float a = bqkv[t];
    for (int i = 0; i < 64; ++i) a += row[i] * wqkv[i*192 + t];
    qkv[(size_t)m*192 + t] = a;
}

__global__ __launch_bounds__(256) void attn_kernel(
    const float* __restrict__ qkv, const unsigned char* __restrict__ mask,
    float* __restrict__ attn)
{
    __shared__ float Ks[E_*HD_];
    __shared__ float Vs[E_*HD_];
    __shared__ float ms[E_];
    const int b = blockIdx.x >> 2, h = blockIdx.x & 3, t = threadIdx.x;
    for (int idx = t; idx < E_*HD_; idx += 256) {
        const int e = idx >> 4, d = idx & 15;
        const size_t base = ((size_t)(b*E_ + e))*192 + h*HD_ + d;
        Ks[idx] = qkv[base + 64];
        Vs[idx] = qkv[base + 128];
    }
    for (int idx = t; idx < E_; idx += 256)
        ms[idx] = mask[b*E_ + idx] ? 1.0f : 0.0f;
    __syncthreads();
    if (t < E_) {
        float q[HD_];
        const size_t qb = ((size_t)(b*E_ + t))*192 + h*HD_;
        #pragma unroll
        for (int d = 0; d < HD_; ++d) q[d] = qkv[qb + d];
        float mx = -3.0e38f, l = 0.0f;
        float o[HD_];
        #pragma unroll
        for (int d = 0; d < HD_; ++d) o[d] = 0.0f;
        for (int kk = 0; kk < E_; ++kk) {
            float s = 0.0f;
            #pragma unroll
            for (int d = 0; d < HD_; ++d) s += q[d] * Ks[kk*HD_ + d];
            s = (ms[kk] != 0.0f) ? s * 0.25f : -1e9f;     // scale = 1/sqrt(16)
            const float nm   = fmaxf(mx, s);
            const float corr = expf(mx - nm);
            const float w    = expf(s - nm);
            l = l * corr + w;
            #pragma unroll
            for (int d = 0; d < HD_; ++d) o[d] = o[d]*corr + w * Vs[kk*HD_ + d];
            mx = nm;
        }
        const float inv = 1.0f / l;
        float* op = attn + (((size_t)(b*H_ + h)*E_ + t)*HD_);
        #pragma unroll
        for (int d = 0; d < HD_; ++d) op[d] = o[d] * inv;
    }
}

__global__ __launch_bounds__(64) void oproj_kernel(
    const float* __restrict__ attn, const float* __restrict__ wo,
    const float* __restrict__ bo, const float* __restrict__ g,
    const float* __restrict__ bln, float* __restrict__ emb)
{
    __shared__ float o64[64];
    __shared__ float buf[64];
    const int m = blockIdx.x, t = threadIdx.x;
    const int b = m / E_, e = m % E_;
    const int h = t >> 4, d = t & 15;
    o64[t] = attn[((size_t)(b*H_ + h)*E_ + e)*HD_ + d];
    __syncthreads();
    float a = bo[t];
    for (int j = 0; j < 64; ++j) a += o64[j] * wo[j*64 + t];
    const float res  = emb[(size_t)m*64 + t] + a;
    const float mean = block_reduce_sum(res, buf) * (1.0f/64.0f);
    const float dv   = res - mean;
    const float var  = block_reduce_sum(dv*dv, buf) * (1.0f/64.0f);
    emb[(size_t)m*64 + t] = dv * rsqrtf(var + 1e-5f) * g[t] + bln[t];
}

__global__ __launch_bounds__(256) void ff_kernel(
    const float* __restrict__ w1, const float* __restrict__ b1,
    const float* __restrict__ w2, const float* __restrict__ b2,
    const float* __restrict__ g,  const float* __restrict__ bln,
    float* __restrict__ emb)
{
    __shared__ float row[64];
    __shared__ float f1[256];
    __shared__ float buf[256];
    const int m = blockIdx.x, t = threadIdx.x;
    if (t < 64) row[t] = emb[(size_t)m*64 + t];
    __syncthreads();
    float a = b1[t];
    for (int i = 0; i < 64; ++i) a += row[i] * w1[i*256 + t];
    f1[t] = gelu_exact(a);
    __syncthreads();
    float res = 0.0f;
    if (t < 64) {
        float a2 = b2[t];
        for (int i = 0; i < 256; ++i) a2 += f1[i] * w2[i*64 + t];
        res = row[t] + a2;
    }
    const float mean = block_reduce_sum(res, buf) * (1.0f/64.0f);
    const float dv   = (t < 64) ? (res - mean) : 0.0f;
    const float var  = block_reduce_sum(dv*dv, buf) * (1.0f/64.0f);
    if (t < 64)
        emb[(size_t)m*64 + t] = dv * rsqrtf(var + 1e-5f) * g[t] + bln[t];
}

// ============================================================================
// Heads
// ============================================================================
__global__ void edge_head_kernel(const float* __restrict__ emb,
    const float* __restrict__ eh_w, const float* __restrict__ eh_b,
    float* __restrict__ out)
{
    const int idx = blockIdx.x * blockDim.x + threadIdx.x;
    if (idx >= M_*2) return;
    const int m = idx >> 1, j = idx & 1;
    float a = eh_b[j];
    for (int d = 0; d < 64; ++d) a += emb[(size_t)m*64 + d] * eh_w[d*2 + j];
    out[idx] = a;
}

__global__ __launch_bounds__(64) void node_kernel(
    const float* __restrict__ emb, const int* __restrict__ p_ptr,
    const int* __restrict__ x_ptr, const int* __restrict__ y_ptr,
    const float* __restrict__ nm_w, const float* __restrict__ nm_b,
    const float* __restrict__ nm_g, const float* __restrict__ nm_bt,
    const float* __restrict__ nh_w, const float* __restrict__ nh_b,
    float* __restrict__ out)
{
    __shared__ float gcat[256];
    __shared__ float nodev[64];
    __shared__ float buf[64];
    const int bu = blockIdx.x;
    const int t  = threadIdx.x;
    const int p  = p_ptr[0], xi = x_ptr[0], yi = y_ptr[0];
    const int b  = bu / 14, u = bu % 14;
    int other = 0, cnt = 0;
    for (int i = 0; i < p; ++i)
        if (i != xi && i != yi) { if (cnt == u) other = i; ++cnt; }
    const int i0 = other*(p-1) + xi - (xi > other ? 1 : 0);
    const int i1 = other*(p-1) + yi - (yi > other ? 1 : 0);
    const int i2 = xi*(p-1) + other - (other > xi ? 1 : 0);
    const int i3 = yi*(p-1) + other - (other > yi ? 1 : 0);
    gcat[t]       = emb[((size_t)b*E_ + i0)*64 + t];
    gcat[64 + t]  = emb[((size_t)b*E_ + i1)*64 + t];
    gcat[128 + t] = emb[((size_t)b*E_ + i2)*64 + t];
    gcat[192 + t] = emb[((size_t)b*E_ + i3)*64 + t];
    __syncthreads();
    float a = nm_b[t];
    for (int j = 0; j < 256; ++j) a += gcat[j] * nm_w[j*64 + t];
    const float mean = block_reduce_sum(a, buf) * (1.0f/64.0f);
    const float dv   = a - mean;
    const float var  = block_reduce_sum(dv*dv, buf) * (1.0f/64.0f);
    nodev[t] = gelu_exact(dv * rsqrtf(var + 1e-5f) * nm_g[t] + nm_bt[t]);
    __syncthreads();
    if (t < 8) {
        float a2 = nh_b[t];
        for (int d = 0; d < 64; ++d) a2 += nodev[d] * nh_w[d*8 + t];
        out[M_*2 + ((size_t)b*14 + u)*8 + t] = a2;
    }
}

// ============================================================================
extern "C" void kernel_launch(void* const* d_in, const int* in_sizes, int n_in,
                              void* d_out, int out_size, void* d_ws, size_t ws_size,
                              hipStream_t stream)
{
    const float* edge_data       = (const float*)d_in[0];
    const int*   edge_types      = (const int*)d_in[1];
    const unsigned char* edge_mask = (const unsigned char*)d_in[2];
    const int*   p_ptr  = (const int*)d_in[3];
    const int*   x_ptr  = (const int*)d_in[4];
    const int*   y_ptr  = (const int*)d_in[5];
    const float* stem_w = (const float*)d_in[6];
    const float* stem_b = (const float*)d_in[7];
    const float* conv_w = (const float*)d_in[8];
    const float* gn_g   = (const float*)d_in[9];
    const float* gn_b   = (const float*)d_in[10];
    const float* type_emb = (const float*)d_in[11];
    const float* em_w  = (const float*)d_in[12];
    const float* em_b  = (const float*)d_in[13];
    const float* em_g  = (const float*)d_in[14];
    const float* em_bt = (const float*)d_in[15];
    const float* a_wqkv = (const float*)d_in[16];
    const float* a_bqkv = (const float*)d_in[17];
    const float* a_wo  = (const float*)d_in[18];
    const float* a_bo  = (const float*)d_in[19];
    const float* ln1_g = (const float*)d_in[20];
    const float* ln1_b = (const float*)d_in[21];
    const float* ff_w1 = (const float*)d_in[22];
    const float* ff_b1 = (const float*)d_in[23];
    const float* ff_w2 = (const float*)d_in[24];
    const float* ff_b2 = (const float*)d_in[25];
    const float* ln2_g = (const float*)d_in[26];
    const float* ln2_b = (const float*)d_in[27];
    const float* eh_w  = (const float*)d_in[28];
    const float* eh_b  = (const float*)d_in[29];
    const float* nm_w  = (const float*)d_in[30];
    const float* nm_b  = (const float*)d_in[31];
    const float* nm_g  = (const float*)d_in[32];
    const float* nm_bt = (const float*)d_in[33];
    const float* nh_w  = (const float*)d_in[34];
    const float* nh_b  = (const float*)d_in[35];

    float* emb_ws  = (float*)d_ws;                    // [1920][64]
    float* qkv_ws  = emb_ws + (size_t)M_*64;          // [1920][192]
    float* attn_ws = qkv_ws + (size_t)M_*192;         // [8][4][240][16]
    float* out = (float*)d_out;

    conv_stack_kernel<<<M_, 256, CONV_LDS_BYTES, stream>>>(
        edge_data, stem_w, stem_b, conv_w, gn_g, gn_b, emb_ws);
    merge_kernel<<<M_, 64, 0, stream>>>(edge_types, type_emb, em_w, em_b, em_g, em_bt, emb_ws);
    for (int l = 0; l < NL_; ++l) {
        qkv_kernel<<<M_, 192, 0, stream>>>(emb_ws, a_wqkv + l*64*192, a_bqkv + l*192, qkv_ws);
        attn_kernel<<<B_*H_, 256, 0, stream>>>(qkv_ws, edge_mask, attn_ws);
        oproj_kernel<<<M_, 64, 0, stream>>>(attn_ws, a_wo + l*64*64, a_bo + l*64,
                                            ln1_g + l*64, ln1_b + l*64, emb_ws);
        ff_kernel<<<M_, 256, 0, stream>>>(ff_w1 + l*64*256, ff_b1 + l*256,
                                          ff_w2 + l*256*64, ff_b2 + l*64,
                                          ln2_g + l*64, ln2_b + l*64, emb_ws);
    }
    edge_head_kernel<<<(M_*2 + 255)/256, 256, 0, stream>>>(emb_ws, eh_w, eh_b, out);
    node_kernel<<<B_*14, 64, 0, stream>>>(emb_ws, p_ptr, x_ptr, y_ptr,
                                          nm_w, nm_b, nm_g, nm_bt, nh_w, nh_b, out);
}
